// GMNN_56831007260969
// MI455X (gfx1250) — matrix operand order repaired and checked by reading, not compiled, not run on hardware
//
#include <hip/hip_runtime.h>
#include <hip/hip_bf16.h>
#include <math.h>

#define N_ATOMS   10000
#define N_PAIRS   600000
#define N_BASIS   7
#define N_RADIAL  5
#define N_SPECIES 119
#define HDIM      512
#define N_GM      360
#define KPAD0     384          // 360 padded to multiple of 32
#define MOM_PER_ATOM 100       // 5 radial * (1 + 3 + 6sym + 10sym)
#define MBLK      32           // atoms per workgroup in MLP kernels

typedef _Float16 f16;
typedef __attribute__((ext_vector_type(16))) _Float16 v16h;
typedef __attribute__((ext_vector_type(8)))  _Float16 v8h;
typedef __attribute__((ext_vector_type(8)))  float    v8f;
typedef __attribute__((ext_vector_type(4)))  int      v4i;
typedef __attribute__((address_space(1))) v4i glb_v4i;   // global
typedef __attribute__((address_space(3))) v4i lds_v4i;   // LDS

#if __has_builtin(__builtin_amdgcn_global_load_async_to_lds_b128)
#define HAVE_ASYNC_LDS 1
#else
#define HAVE_ASYNC_LDS 0
#endif

// ---------------- constant index tables ----------------
__constant__ int TRIL2_I[15] = {0,0,0,0,0,1,1,1,1,2,2,2,3,3,4};
__constant__ int TRIL2_J[15] = {0,1,2,3,4,1,2,3,4,2,3,4,3,4,4};
__constant__ int TRIL3_I[35] = {0,0,0,0,0,0,0,0,0,0,0,0,0,0,0,
                                1,1,1,1,1,1,1,1,1,1,
                                2,2,2,2,2,2,
                                3,3,3,4};
__constant__ int TRIL3_J[35] = {0,0,0,0,0,1,1,1,1,2,2,2,3,3,4,
                                1,1,1,1,2,2,2,3,3,4,
                                2,2,2,3,3,4,
                                3,3,4,4};
__constant__ int TRIL3_K[35] = {0,1,2,3,4,1,2,3,4,2,3,4,3,4,4,
                                1,2,3,4,2,3,4,3,4,4,
                                2,3,4,3,4,4,
                                3,4,4,4};
// symmetric rank-2: (i,j) -> slot in [4..9]  (xx,yy,zz,xy,xz,yz)
__constant__ int S2TAB[9] = {0,3,4, 3,1,5, 4,5,2};
// symmetric rank-3: (i,j,k) -> slot in [10..19] (xxx,xxy,xxz,xyy,xyz,xzz,yyy,yyz,yzz,zzz)
__constant__ int T3TAB[27] = {0,1,2, 1,3,4, 2,4,5,
                              1,3,4, 3,6,7, 4,7,8,
                              2,4,5, 4,7,8, 5,8,9};

// ---------------- phase 1: pair kernel (scatter moments) ----------------
__global__ __launch_bounds__(256) void pair_kernel(
    const float* __restrict__ R, const int* __restrict__ Z,
    const int* __restrict__ idx, const float* __restrict__ Wrad,
    float* __restrict__ mom)
{
  int p = blockIdx.x * blockDim.x + threadIdx.x;
  if (p >= N_PAIRS) return;
  int ia = idx[p];
  int ja = idx[N_PAIRS + p];
  float dx = R[3*ja+0] - R[3*ia+0];
  float dy = R[3*ja+1] - R[3*ia+1];
  float dz = R[3*ja+2] - R[3*ia+2];
  float dr = sqrtf(dx*dx + dy*dy + dz*dz);
  float drs = dr > 1e-6f ? dr : 1e-6f;
  float inv = 1.0f / drs;
  float nx = dx*inv, ny = dy*inv, nz = dz*inv;

  const float betta    = (float)(N_BASIS*N_BASIS) / 36.0f;   // 49/36
  const float rad_norm = 0.96481705f;                        // (2*betta/pi)^0.25
  const float embed_nm = 0.37796447f;                        // 1/sqrt(7)

  float basis[N_BASIS];
#pragma unroll
  for (int b = 0; b < N_BASIS; ++b) {
    float sh = 0.5f + (5.5f / 7.0f) * (float)b;
    float d  = dr - sh;
    basis[b] = rad_norm * __expf(-betta * d * d);
  }
  float cutoff = (dr < 6.0f) ? 0.5f * (__cosf(0.52359878f * dr) + 1.0f) : 0.0f;

  int Zi = Z[ia], Zj = Z[ja];
  float radial[N_RADIAL];
#pragma unroll
  for (int r = 0; r < N_RADIAL; ++r) {
    const float* c = Wrad + ((size_t)((r * N_SPECIES + Zi) * N_SPECIES + Zj)) * N_BASIS;
    float s = 0.0f;
#pragma unroll
    for (int b = 0; b < N_BASIS; ++b) s += c[b] * basis[b];
    radial[r] = embed_nm * s * cutoff;
  }

  float* mi = mom + (size_t)ia * MOM_PER_ATOM;
#pragma unroll
  for (int r = 0; r < N_RADIAL; ++r) {
    float v = radial[r];
    float* mr = mi + r * 20;
    atomicAdd(mr + 0,  v);
    atomicAdd(mr + 1,  v * nx);
    atomicAdd(mr + 2,  v * ny);
    atomicAdd(mr + 3,  v * nz);
    atomicAdd(mr + 4,  v * nx * nx);
    atomicAdd(mr + 5,  v * ny * ny);
    atomicAdd(mr + 6,  v * nz * nz);
    atomicAdd(mr + 7,  v * nx * ny);
    atomicAdd(mr + 8,  v * nx * nz);
    atomicAdd(mr + 9,  v * ny * nz);
    atomicAdd(mr + 10, v * nx * nx * nx);
    atomicAdd(mr + 11, v * nx * nx * ny);
    atomicAdd(mr + 12, v * nx * nx * nz);
    atomicAdd(mr + 13, v * nx * ny * ny);
    atomicAdd(mr + 14, v * nx * ny * nz);
    atomicAdd(mr + 15, v * nx * nz * nz);
    atomicAdd(mr + 16, v * ny * ny * ny);
    atomicAdd(mr + 17, v * ny * ny * nz);
    atomicAdd(mr + 18, v * ny * nz * nz);
    atomicAdd(mr + 19, v * nz * nz * nz);
  }
}

// ---------------- phase 2: per-atom invariant contractions -> gm ----------------
__device__ __forceinline__ float M0(const float* s, int r)               { return s[r*20]; }
__device__ __forceinline__ float M1(const float* s, int r, int i)        { return s[r*20 + 1 + i]; }
__device__ __forceinline__ float M2(const float* s, int r, int i, int j) { return s[r*20 + 4 + S2TAB[3*i + j]]; }
__device__ __forceinline__ float M3(const float* s, int r, int i, int j, int k)
                                                                         { return s[r*20 + 10 + T3TAB[(3*i + j)*3 + k]]; }

__global__ __launch_bounds__(KPAD0) void gm_kernel(
    const float* __restrict__ mom, f16* __restrict__ gm)
{
  __shared__ float sm[MOM_PER_ATOM];
  int a = blockIdx.x;
  int g = threadIdx.x;
  if (g < MOM_PER_ATOM) sm[g] = mom[(size_t)a * MOM_PER_ATOM + g];
  __syncthreads();

  if (g >= N_GM) { gm[(size_t)a * KPAD0 + g] = (f16)0.0f; return; }

  float val = 0.0f;
  if (g < 5) {
    val = M0(sm, g);
  } else if (g < 20) {                                // c1
    int q = g - 5, r = TRIL2_I[q], s = TRIL2_J[q];
    for (int i = 0; i < 3; ++i) val += M1(sm,r,i) * M1(sm,s,i);
  } else if (g < 35) {                                // c2
    int q = g - 20, r = TRIL2_I[q], s = TRIL2_J[q];
    for (int i = 0; i < 3; ++i) for (int j = 0; j < 3; ++j)
      val += M2(sm,r,i,j) * M2(sm,s,i,j);
  } else if (g < 50) {                                // c3
    int q = g - 35, r = TRIL2_I[q], s = TRIL2_J[q];
    for (int i = 0; i < 3; ++i) for (int j = 0; j < 3; ++j) for (int k = 0; k < 3; ++k)
      val += M3(sm,r,i,j,k) * M3(sm,s,i,j,k);
  } else if (g < 85) {                                // c4
    int q = g - 50, r = TRIL3_I[q], s = TRIL3_J[q], t = TRIL3_K[q];
    for (int i = 0; i < 3; ++i) for (int j = 0; j < 3; ++j) for (int k = 0; k < 3; ++k)
      val += M2(sm,r,i,j) * M2(sm,s,i,k) * M2(sm,t,j,k);
  } else if (g < 160) {                               // c5[:, i2, j2]
    int q = g - 85, pr = q / 5, t = q % 5;
    int r = TRIL2_I[pr], s = TRIL2_J[pr];
    for (int i = 0; i < 3; ++i) for (int j = 0; j < 3; ++j)
      val += M1(sm,r,i) * M1(sm,s,j) * M2(sm,t,i,j);
  } else if (g < 235) {                               // c6
    int q = g - 160, pr = q / 5, t = q % 5;
    int r = TRIL2_I[pr], s = TRIL2_J[pr];
    for (int i = 0; i < 3; ++i) for (int j = 0; j < 3; ++j)
      for (int k = 0; k < 3; ++k) for (int l = 0; l < 3; ++l)
        val += M3(sm,r,i,j,k) * M3(sm,s,i,j,l) * M2(sm,t,k,l);
  } else {                                            // c7 full 5x5x5
    int q = g - 235, r = q / 25, s = (q / 5) % 5, t = q % 5;
    for (int i = 0; i < 3; ++i) for (int j = 0; j < 3; ++j) for (int k = 0; k < 3; ++k)
      val += M3(sm,r,i,j,k) * M2(sm,s,i,j) * M1(sm,t,k);
  }
  gm[(size_t)a * KPAD0 + g] = (f16)val;
}

// ---------------- weight prepack: f32 [K][N] -> f16 [N][Kpad] (n-major) ----------------
__global__ __launch_bounds__(256) void prepack_w0(const float* __restrict__ W, f16* __restrict__ Wt)
{
  int t = blockIdx.x * blockDim.x + threadIdx.x;
  if (t >= HDIM * KPAD0) return;
  int n = t / KPAD0, k = t % KPAD0;
  Wt[t] = (k < N_GM) ? (f16)W[(size_t)k * HDIM + n] : (f16)0.0f;
}
__global__ __launch_bounds__(256) void prepack_w1(const float* __restrict__ W, f16* __restrict__ Wt)
{
  int t = blockIdx.x * blockDim.x + threadIdx.x;
  if (t >= HDIM * HDIM) return;
  int n = t / HDIM, k = t % HDIM;
  Wt[t] = (f16)W[(size_t)k * HDIM + n];
}

// ---------------- phase 3: WMMA MLP layer ----------------
// Block = 256 threads (8 waves). One block: MBLK(=32) atoms x 512 outputs.
// A slab (32 x LDA f16) staged in LDS once per block via async global->LDS
// (ASYNCcnt path) with +8-half row padding for conflict-free ds_load_b128.
// Wave w owns columns [64w,64w+64) as 4 N-tiles; 2 M-tiles -> 8 accumulators.
// B fragments are double-buffered so L2 latency overlaps the WMMAs.
template <int KTILES, int LDA>
__global__ __launch_bounds__(256) void mlp_layer(
    const f16* __restrict__ A, const f16* __restrict__ Bt,
    const float* __restrict__ bias, float inv_sqrt_k,
    f16* __restrict__ Out)
{
  constexpr int LDS_LDA = LDA + 8;                 // pad 16B -> stride 4 banks/row
  __shared__ f16 smA[MBLK * LDS_LDA];

  int tid  = threadIdx.x;
  int wave = tid >> 5;
  int lane = tid & 31;
  int lh   = lane & 15;
  int grp  = lane >> 4;
  int m0   = blockIdx.x * MBLK;
  int n0w  = wave * 64;

  // ---- stage A slab into LDS (16B chunks), clamp rows for the tail block ----
  constexpr int CHUNKS_PER_ROW = LDA / 8;
  for (int c = tid; c < MBLK * CHUNKS_PER_ROW; c += 256) {
    int r  = c / CHUNKS_PER_ROW;
    int k8 = (c % CHUNKS_PER_ROW) * 8;
    int gr = m0 + r; if (gr >= N_ATOMS) gr = N_ATOMS - 1;
    const f16* gsrc = A + (size_t)gr * LDA + k8;
    f16*       ldst = smA + r * LDS_LDA + k8;
#if HAVE_ASYNC_LDS
    __builtin_amdgcn_global_load_async_to_lds_b128(
        (glb_v4i*)gsrc, (lds_v4i*)ldst, 0, 0);
#else
    *(v8h*)ldst = *(const v8h*)gsrc;
#endif
  }
#if HAVE_ASYNC_LDS
#if __has_builtin(__builtin_amdgcn_s_wait_asynccnt)
  __builtin_amdgcn_s_wait_asynccnt(0);
#else
  asm volatile("s_wait_asynccnt 0x0" ::: "memory");
#endif
#endif
  __syncthreads();

  const f16* brow = Bt + (size_t)(n0w + lh) * LDA + 16 * grp;

  v8f  acc[2][4] = {};
  v16h bcur[4], bnext[4];
#pragma unroll
  for (int t = 0; t < 4; ++t) bcur[t] = *(const v16h*)(brow + (size_t)t * 16 * LDA);

#pragma unroll
  for (int kt = 0; kt < KTILES; ++kt) {
    int kb = kt * 32;
    if (kt + 1 < KTILES) {
#pragma unroll
      for (int t = 0; t < 4; ++t)
        bnext[t] = *(const v16h*)(brow + (kt + 1) * 32 + (size_t)t * 16 * LDA);
    }
#pragma unroll
    for (int mt = 0; mt < 2; ++mt) {
      const f16* ar = smA + (mt * 16 + lh) * LDS_LDA + kb + 16 * grp;
      v8h alo = *(const v8h*)ar;
      v8h ahi = *(const v8h*)(ar + 8);
      v16h av;
#pragma unroll
      for (int e = 0; e < 8; ++e) { av[e] = alo[e]; av[e + 8] = ahi[e]; }
#pragma unroll
      for (int t = 0; t < 4; ++t)
        acc[mt][t] = __builtin_amdgcn_wmma_f32_16x16x32_f16(
            false, av, false, bcur[t], (short)0, acc[mt][t], false, false);
    }
#pragma unroll
    for (int t = 0; t < 4; ++t) bcur[t] = bnext[t];
  }

  // ---- epilogue: ntk scale + bias, swish, f16 store (predicated on tail) ----
#pragma unroll
  for (int mt = 0; mt < 2; ++mt) {
#pragma unroll
    for (int t = 0; t < 4; ++t) {
      int n = n0w + 16 * t + lh;
      float bb = 0.1f * bias[n];
#pragma unroll
      for (int v = 0; v < 8; ++v) {
        int row = m0 + mt * 16 + v + 8 * grp;    // C layout: vgpr v -> M = v + 8*grp
        float y = acc[mt][t][v] * inv_sqrt_k + bb;
        float h = y / (1.0f + __expf(-y));       // swish
        if (row < N_ATOMS) Out[(size_t)row * HDIM + n] = (f16)h;
      }
    }
  }
}

// ---------------- head: 512 -> 1 + species scale/shift ----------------
__global__ __launch_bounds__(256) void head_kernel(
    const f16* __restrict__ h1, const float* __restrict__ W2,
    const float* __restrict__ b2, const float* __restrict__ scale,
    const float* __restrict__ shift, const int* __restrict__ Z,
    float* __restrict__ out)
{
  int a = blockIdx.x * blockDim.x + threadIdx.x;
  if (a >= N_ATOMS) return;
  const f16* row = h1 + (size_t)a * HDIM;
  float s = 0.0f;
#pragma unroll 4
  for (int k = 0; k < HDIM; ++k) s += (float)row[k] * W2[k];
  float y = s * 0.044194174f + 0.1f * b2[0];      // 1/sqrt(512)
  int z = Z[a];
  float o = scale[z] * y + shift[z];
  out[a] = (z != 0) ? o : 0.0f;
}

// ---------------- launcher ----------------
extern "C" void kernel_launch(void* const* d_in, const int* in_sizes, int n_in,
                              void* d_out, int out_size, void* d_ws, size_t ws_size,
                              hipStream_t stream)
{
  (void)in_sizes; (void)n_in; (void)out_size; (void)ws_size;
  const float* R     = (const float*)d_in[0];
  const int*   Z     = (const int*)  d_in[1];
  const int*   idx   = (const int*)  d_in[2];
  const float* Wrad  = (const float*)d_in[3];
  const float* W0    = (const float*)d_in[4];
  const float* b0    = (const float*)d_in[5];
  const float* W1    = (const float*)d_in[6];
  const float* b1    = (const float*)d_in[7];
  const float* W2    = (const float*)d_in[8];
  const float* b2    = (const float*)d_in[9];
  const float* scale = (const float*)d_in[10];
  const float* shift = (const float*)d_in[11];
  float* out = (float*)d_out;

  char* ws = (char*)d_ws;
  size_t off = 0;
  auto carve = [&](size_t bytes) { void* p = ws + off; off += (bytes + 255) & ~(size_t)255; return p; };
  float* mom = (float*)carve((size_t)N_ATOMS * MOM_PER_ATOM * sizeof(float));
  f16*   gm  = (f16*)  carve((size_t)N_ATOMS * KPAD0 * sizeof(f16));
  f16*   h0  = (f16*)  carve((size_t)N_ATOMS * HDIM * sizeof(f16));
  f16*   h1  = (f16*)  carve((size_t)N_ATOMS * HDIM * sizeof(f16));
  f16*   Wt0 = (f16*)  carve((size_t)HDIM * KPAD0 * sizeof(f16));
  f16*   Wt1 = (f16*)  carve((size_t)HDIM * HDIM * sizeof(f16));

  (void)hipMemsetAsync(mom, 0, (size_t)N_ATOMS * MOM_PER_ATOM * sizeof(float), stream);

  prepack_w0<<<(HDIM * KPAD0 + 255) / 256, 256, 0, stream>>>(W0, Wt0);
  prepack_w1<<<(HDIM * HDIM + 255) / 256, 256, 0, stream>>>(W1, Wt1);

  pair_kernel<<<(N_PAIRS + 255) / 256, 256, 0, stream>>>(R, Z, idx, Wrad, mom);
  gm_kernel<<<N_ATOMS, KPAD0, 0, stream>>>(mom, gm);

  int mlp_blocks = (N_ATOMS + MBLK - 1) / MBLK;   // 313
  mlp_layer<12, KPAD0><<<mlp_blocks, 256, 0, stream>>>(gm, Wt0, b0, 0.05270463f /*1/sqrt(360)*/, h0);
  mlp_layer<16, HDIM ><<<mlp_blocks, 256, 0, stream>>>(h0, Wt1, b1, 0.04419417f /*1/sqrt(512)*/, h1);

  head_kernel<<<(N_ATOMS + 255) / 256, 256, 0, stream>>>(h1, W2, b2, scale, shift, Z, out);
}
